// GraphSAGEModel_49675591745699
// MI455X (gfx1250) — matrix operand order repaired and checked by reading, not compiled, not run on hardware
//
#include <hip/hip_runtime.h>
#include <hip/hip_bf16.h>

#define NNODES 100000
#define HDIM   128

typedef __attribute__((ext_vector_type(16))) __bf16 v16bf;
typedef __attribute__((ext_vector_type(8)))  float  v8f;

// ---------------------------------------------------------------------------
// A-fragment: 16x32 bf16, MxK. k order used here (contiguous, k = 16h + s; B matches):
// lanes 0-15 : row M=lane,    VGPR0..7 -> K=0..15
// lanes 16-31: row M=lane-16, VGPR0..7 -> K=16..31
// Source is f32 row-major (ld = HDIM); optional per-row scale (mean divide).
__device__ inline v16bf load_a_frag(const float* __restrict__ A, int row0,
                                    int kbase, const float* __restrict__ inv,
                                    int lane) {
  int row   = row0 + (lane & 15);
  int khalf = lane >> 4;
  float sc  = inv ? inv[row] : 1.0f;
  const float* p = A + (size_t)row * HDIM + kbase;
  v16bf a;
#pragma unroll
  for (int v = 0; v < 8; ++v) {
    int k0 = 2 * v + khalf * 16;
    a[2 * v]     = (__bf16)(p[k0]     * sc);
    a[2 * v + 1] = (__bf16)(p[k0 + 1] * sc);
  }
  return a;
}

// B-fragment: 32x16 bf16, KxN. lanes 0-15 col N=lane (K=0..15 over 8 VGPRs),
// lanes 16-31 col N=lane-16 (K=16..31). Weights are pre-transposed to
// bf16 [N][K] so each lane's 16 K-values are one contiguous 32B vector.
__device__ inline v16bf load_b_frag(const __bf16* __restrict__ Wt, int kbase,
                                    int col0, int lane) {
  int col   = col0 + (lane & 15);
  int khalf = lane >> 4;
  return *(const v16bf*)(Wt + (size_t)col * HDIM + kbase + khalf * 16);
}

// Out[N,128] = (Aagg*inv) @ Wl + bias (+ Ax @ Wr if Ax).  One wave = one
// 16x16 tile; block = 8 waves = 16 rows x all 128 cols; grid = N/16.
__global__ void sage_gemm_wmma(const float* __restrict__ Aagg,
                               const float* __restrict__ inv,
                               const float* __restrict__ Ax,
                               const __bf16* __restrict__ Wl,
                               const __bf16* __restrict__ Wr,
                               const float* __restrict__ bias,
                               float* __restrict__ Out) {
  int lane = threadIdx.x & 31;
  int col0 = (threadIdx.x >> 5) * 16;
  int row0 = blockIdx.x * 16;
  v8f c = {};
#pragma unroll
  for (int kb = 0; kb < HDIM; kb += 32) {
    v16bf a = load_a_frag(Aagg, row0, kb, inv, lane);
    v16bf b = load_b_frag(Wl, kb, col0, lane);
    c = __builtin_amdgcn_wmma_f32_16x16x32_bf16(false, a, false, b,
                                                (short)0, c, false, false);
  }
  if (Ax) {
#pragma unroll
    for (int kb = 0; kb < HDIM; kb += 32) {
      v16bf a = load_a_frag(Ax, row0, kb, nullptr, lane);
      v16bf b = load_b_frag(Wr, kb, col0, lane);
      c = __builtin_amdgcn_wmma_f32_16x16x32_bf16(false, a, false, b,
                                                  (short)0, c, false, false);
    }
  }
  int col = col0 + (lane & 15);
  float bv = bias ? bias[col] : 0.0f;
  int mbase = (lane >> 4) ? 8 : 0;  // C/D layout: VGPR v -> M=v or v+8
#pragma unroll
  for (int v = 0; v < 8; ++v)
    Out[(size_t)(row0 + mbase + v) * HDIM + col] = c[v] + bv;
}

// ---------------------------------------------------------------------------
__global__ void zero_b32(unsigned* __restrict__ p, long long n) {
  long long i = (long long)blockIdx.x * blockDim.x + threadIdx.x;
  if (i < n) p[i] = 0u;
}

// weight convert + transpose: W f32 [K][N] -> Wt bf16 [N][K]
__global__ void wt_bf16(const float* __restrict__ W, __bf16* __restrict__ Wt) {
  int k = blockIdx.x;    // 0..127
  int n = threadIdx.x;   // 0..127
  Wt[(size_t)n * HDIM + k] = (__bf16)W[(size_t)k * HDIM + n];
}

__global__ void count_edges(const long long* __restrict__ dst, long long E,
                            unsigned* __restrict__ cnt) {
  long long e = (long long)blockIdx.x * 256 + threadIdx.x;
  if (e < E) atomicAdd(&cnt[dst[e]], 1u);
}

__global__ void make_inv(const unsigned* __restrict__ cnt, float* __restrict__ inv) {
  int i = blockIdx.x * 256 + threadIdx.x;
  if (i < NNODES) inv[i] = 1.0f / fmaxf((float)cnt[i], 1.0f);
}

// one wave per edge: lane l moves features [4l, 4l+4) with hw fp32 atomics (L2 RMW)
__global__ void scatter_add_edges(const float* __restrict__ X,
                                  const long long* __restrict__ src,
                                  const long long* __restrict__ dst,
                                  long long E, float* __restrict__ Agg) {
  long long e = (long long)blockIdx.x * 8 + (threadIdx.x >> 5);
  if (e >= E) return;
  int lane = threadIdx.x & 31;
  long long s = src[e], d = dst[e];
  float4 v = *(const float4*)(X + s * HDIM + lane * 4);
  float* o = Agg + d * HDIM + lane * 4;
  unsafeAtomicAdd(o + 0, v.x);
  unsafeAtomicAdd(o + 1, v.y);
  unsafeAtomicAdd(o + 2, v.z);
  unsafeAtomicAdd(o + 3, v.w);
}

// column sums / sums-of-squares over the node dim (LDS pair-reduce + atomics)
__global__ void col_stats(const float* __restrict__ T, float* __restrict__ sums,
                          float* __restrict__ sqs) {
  __shared__ float ls[HDIM], lq[HDIM];
  int col  = threadIdx.x & (HDIM - 1);
  int half = threadIdx.x >> 7;
  float s = 0.0f, q = 0.0f;
  for (int row = blockIdx.x * 2 + half; row < NNODES; row += gridDim.x * 2) {
    float v = T[(size_t)row * HDIM + col];
    s += v; q += v * v;
  }
  if (half == 0) { ls[col] = s; lq[col] = q; }
  __syncthreads();
  if (half == 1) { ls[col] += s; lq[col] += q; }
  __syncthreads();
  if (threadIdx.x < HDIM) {
    unsafeAtomicAdd(&sums[col], ls[col]);
    unsafeAtomicAdd(&sqs[col],  lq[col]);
  }
}

__global__ void finalize_stats(const float* __restrict__ sums,
                               const float* __restrict__ sqs,
                               float* __restrict__ mu, float* __restrict__ rstd) {
  int c = threadIdx.x;
  if (c < HDIM) {
    float m = sums[c] * (1.0f / NNODES);
    float v = sqs[c] * (1.0f / NNODES) - m * m;
    mu[c] = m;
    rstd[c] = rsqrtf(fmaxf(v, 0.0f) + 1e-5f);
  }
}

// x_next = elu(gamma*(t-mu)*rstd + beta) [+ R]
__global__ void bn_elu_res(const float* __restrict__ T, const float* __restrict__ mu,
                           const float* __restrict__ rstd, const float* __restrict__ g,
                           const float* __restrict__ be, const float* __restrict__ R,
                           float* __restrict__ Out, long long total) {
  long long i = (long long)blockIdx.x * 256 + threadIdx.x;
  if (i >= total) return;
  int col = (int)(i & (HDIM - 1));
  float v = g[col] * (T[i] - mu[col]) * rstd[col] + be[col];
  v = v > 0.0f ? v : expm1f(v);
  if (R) v += R[i];
  Out[i] = v;
}

// layer 4 (H -> 1): out[n] = sum_k agg[n,k]*inv[n]*W4l[k] + x[n,k]*W4r[k] + b4
__global__ void final_sage(const float* __restrict__ Agg, const float* __restrict__ inv,
                           const float* __restrict__ X, const float* __restrict__ W4l,
                           const float* __restrict__ b4, const float* __restrict__ W4r,
                           float* __restrict__ Out) {
  int n = blockIdx.x * 8 + (threadIdx.x >> 5);
  if (n >= NNODES) return;
  int lane = threadIdx.x & 31;
  float iv = inv[n], s = 0.0f;
#pragma unroll
  for (int j = 0; j < 4; ++j) {
    int k = lane * 4 + j;
    s += Agg[(size_t)n * HDIM + k] * iv * W4l[k] + X[(size_t)n * HDIM + k] * W4r[k];
  }
  for (int m = 16; m >= 1; m >>= 1) s += __shfl_xor(s, m, 32);
  if (lane == 0) Out[n] = s + b4[0];
}

// ---------------------------------------------------------------------------
extern "C" void kernel_launch(void* const* d_in, const int* in_sizes, int n_in,
                              void* d_out, int out_size, void* d_ws, size_t ws_size,
                              hipStream_t stream) {
  const float*     x    = (const float*)d_in[0];
  const long long* ei   = (const long long*)d_in[1];
  const long long  E    = (long long)in_sizes[1] / 2;
  const long long* src  = ei;
  const long long* dst  = ei + E;
  const float *W1l = (const float*)d_in[2],  *b1 = (const float*)d_in[3],
              *W1r = (const float*)d_in[4],  *W2l = (const float*)d_in[5],
              *b2  = (const float*)d_in[6],  *W2r = (const float*)d_in[7],
              *W3l = (const float*)d_in[8],  *b3 = (const float*)d_in[9],
              *W3r = (const float*)d_in[10], *W4l = (const float*)d_in[11],
              *b4  = (const float*)d_in[12], *W4r = (const float*)d_in[13],
              *g1  = (const float*)d_in[14], *be1 = (const float*)d_in[15],
              *g2  = (const float*)d_in[16], *be2 = (const float*)d_in[17],
              *g3  = (const float*)d_in[18], *be3 = (const float*)d_in[19],
              *Wln = (const float*)d_in[20], *bln = (const float*)d_in[21];
  float* out = (float*)d_out;

  const size_t NH = (size_t)NNODES * HDIM;  // elements
  char* p = (char*)d_ws;
  float* agg = (float*)p;               p += NH * 4;   // also residual buf
  float* t   = (float*)p;               p += NH * 4;
  float* xa  = (float*)p;               p += NH * 4;
  float* xb  = (float*)p;               p += NH * 4;
  float* inv = (float*)p;               p += (size_t)NNODES * 4;
  unsigned* cnt = (unsigned*)p;         p += (size_t)NNODES * 4;
  __bf16* wbf = (__bf16*)p;             p += (size_t)7 * HDIM * HDIM * 2;
  float* sums = (float*)p;              p += HDIM * 4;
  float* sqs  = (float*)p;              p += HDIM * 4;
  float* mu   = (float*)p;              p += HDIM * 4;
  float* rstd = (float*)p;              p += HDIM * 4;
  __bf16 *w1l = wbf + 0 * HDIM * HDIM, *w1r = wbf + 1 * HDIM * HDIM,
         *w2l = wbf + 2 * HDIM * HDIM, *w2r = wbf + 3 * HDIM * HDIM,
         *w3l = wbf + 4 * HDIM * HDIM, *w3r = wbf + 5 * HDIM * HDIM,
         *wli = wbf + 6 * HDIM * HDIM;

  const int B = 256;
  const int gN  = (NNODES + B - 1) / B;            // node-wise
  const int gNH = (int)((NH + B - 1) / B);         // elementwise N*H
  const int gE  = (int)((E + B - 1) / B);
  const int gEw = (int)((E + 7) / 8);              // wave-per-edge
  const int gT  = NNODES / 16;                     // GEMM row tiles

  // degrees + reciprocal (edge structure is layer-invariant)
  zero_b32<<<gN, B, 0, stream>>>(cnt, NNODES);
  count_edges<<<gE, B, 0, stream>>>(dst, E, cnt);
  make_inv<<<gN, B, 0, stream>>>(cnt, inv);

  // weights -> bf16, transposed to [N][K] for contiguous B-fragment loads
  wt_bf16<<<HDIM, HDIM, 0, stream>>>(W1l, w1l);
  wt_bf16<<<HDIM, HDIM, 0, stream>>>(W1r, w1r);
  wt_bf16<<<HDIM, HDIM, 0, stream>>>(W2l, w2l);
  wt_bf16<<<HDIM, HDIM, 0, stream>>>(W2r, w2r);
  wt_bf16<<<HDIM, HDIM, 0, stream>>>(W3l, w3l);
  wt_bf16<<<HDIM, HDIM, 0, stream>>>(W3r, w3r);
  wt_bf16<<<HDIM, HDIM, 0, stream>>>(Wln, wli);

  // ---- layer 1: xa = elu(bn(sage(x)))
  zero_b32<<<gNH, B, 0, stream>>>((unsigned*)agg, (long long)NH);
  scatter_add_edges<<<gEw, B, 0, stream>>>(x, src, dst, E, agg);
  sage_gemm_wmma<<<gT, B, 0, stream>>>(agg, inv, x, w1l, w1r, b1, t);
  zero_b32<<<1, B, 0, stream>>>((unsigned*)sums, 2 * HDIM);
  col_stats<<<512, B, 0, stream>>>(t, sums, sqs);
  finalize_stats<<<1, HDIM, 0, stream>>>(sums, sqs, mu, rstd);
  bn_elu_res<<<gNH, B, 0, stream>>>(t, mu, rstd, g1, be1, nullptr, xa, (long long)NH);

  // ---- layer 2: xb = elu(bn(sage(xa))) + xa@Wlin + blin
  zero_b32<<<gNH, B, 0, stream>>>((unsigned*)agg, (long long)NH);
  scatter_add_edges<<<gEw, B, 0, stream>>>(xa, src, dst, E, agg);
  sage_gemm_wmma<<<gT, B, 0, stream>>>(agg, inv, xa, w2l, w2r, b2, t);
  sage_gemm_wmma<<<gT, B, 0, stream>>>(xa, nullptr, nullptr, wli, wli, bln, agg); // residual
  zero_b32<<<1, B, 0, stream>>>((unsigned*)sums, 2 * HDIM);
  col_stats<<<512, B, 0, stream>>>(t, sums, sqs);
  finalize_stats<<<1, HDIM, 0, stream>>>(sums, sqs, mu, rstd);
  bn_elu_res<<<gNH, B, 0, stream>>>(t, mu, rstd, g2, be2, agg, xb, (long long)NH);

  // ---- layer 3: xa = elu(bn(sage(xb))) + xb@Wlin + blin
  zero_b32<<<gNH, B, 0, stream>>>((unsigned*)agg, (long long)NH);
  scatter_add_edges<<<gEw, B, 0, stream>>>(xb, src, dst, E, agg);
  sage_gemm_wmma<<<gT, B, 0, stream>>>(agg, inv, xb, w3l, w3r, b3, t);
  sage_gemm_wmma<<<gT, B, 0, stream>>>(xb, nullptr, nullptr, wli, wli, bln, agg); // residual
  zero_b32<<<1, B, 0, stream>>>((unsigned*)sums, 2 * HDIM);
  col_stats<<<512, B, 0, stream>>>(t, sums, sqs);
  finalize_stats<<<1, HDIM, 0, stream>>>(sums, sqs, mu, rstd);
  bn_elu_res<<<gNH, B, 0, stream>>>(t, mu, rstd, g3, be3, agg, xa, (long long)NH);

  // ---- layer 4: out = sage(xa) with H->1
  zero_b32<<<gNH, B, 0, stream>>>((unsigned*)agg, (long long)NH);
  scatter_add_edges<<<gEw, B, 0, stream>>>(xa, src, dst, E, agg);
  final_sage<<<(NNODES + 7) / 8, B, 0, stream>>>(agg, inv, xa, W4l, b4, W4r, out);
}
